// CoordAttention_44324062494802
// MI455X (gfx1250) — matrix operand order repaired and checked by reading, not compiled, not run on hardware
//
#include <hip/hip_runtime.h>
#include <hip/hip_bf16.h>

typedef __attribute__((ext_vector_type(16))) _Float16 v16h;
typedef __attribute__((ext_vector_type(8)))  _Float16 v8h;
typedef __attribute__((ext_vector_type(2)))  _Float16 h2;
typedef __attribute__((ext_vector_type(8)))  float    v8f;

#define NTOK   1536
#define NBATCH 2
#define NHEAD  12
#define HD     32
#define DIM    384
#define QKVN   1152
#define CPBH   32
#define SCALE  0.17677669529663687f   // 32^-0.5

__device__ __forceinline__ h2 splat2(float f) {
  _Float16 t = (_Float16)f;
  h2 v = { t, t };
  return v;
}

// ---------------------------------------------------------------------------
// CDNA5 WMMA fragment loaders (wave32, contiguous k order, same for A and B)
// A 16x32 (MxK): lane L holds row m=L&15; kBase=(L<16)?0:16; half h -> K = kBase + h
// B 32x16 (KxN): lane L holds col n=L&15; kBase=(L<16)?0:16; half h -> K = kBase + h
// C/D 16x16 f32: lane L reg r -> (m = r + 8*(L>>4), n = L&15)
// ---------------------------------------------------------------------------
__device__ __forceinline__ v16h load_a_frag_f16(const _Float16* a, int lda) {
  int lane = threadIdx.x & 31;
  int m  = lane & 15;
  int kb = (lane < 16) ? 0 : 16;
  const _Float16* p = a + (size_t)m * lda + kb;
  v8h lo = *(const v8h*)(p);        // K = kb .. kb+7
  v8h hi = *(const v8h*)(p + 8);    // K = kb+8 .. kb+15
  v16h r;
#pragma unroll
  for (int i = 0; i < 8; ++i) { r[i] = lo[i]; r[8 + i] = hi[i]; }
  return r;
}

// B-fragment where memory holds B^T row-major (N x K): one contiguous 32B load
__device__ __forceinline__ v16h load_b_frag_bt(const _Float16* bt, int ldbt) {
  int lane = threadIdx.x & 31;
  int n  = lane & 15;
  int kb = (lane < 16) ? 0 : 16;
  return *(const v16h*)(bt + (size_t)n * ldbt + kb);
}

__device__ __forceinline__ v8f wmma_f16(v16h a, v16h b, v8f c) {
  return __builtin_amdgcn_wmma_f32_16x16x32_f16(false, a, false, b, (short)0, c, false, false);
}

// ---------------------------------------------------------------------------
// plain f32 -> f16 conversion (for activations)
// ---------------------------------------------------------------------------
__global__ void cvt_f16_kernel(const float* __restrict__ src,
                               _Float16* __restrict__ dst, int n) {
  int i = blockIdx.x * 256 + threadIdx.x;
  if (i < n) dst[i] = (_Float16)src[i];
}

// ---------------------------------------------------------------------------
// f32 (K x N row-major) -> f16 transposed (N x K row-major)
// makes every WMMA B-fragment a single contiguous 32B load
// ---------------------------------------------------------------------------
__global__ void cvt_transpose_f16_kernel(const float* __restrict__ src,
                                         _Float16* __restrict__ dst,
                                         int K, int N) {
  int idx = blockIdx.x * 256 + threadIdx.x;
  if (idx < K * N) {
    int n = idx / K, kk = idx % K;          // dst-linear order: coalesced stores
    dst[idx] = (_Float16)src[(size_t)kk * N + n];
  }
}

// ---------------------------------------------------------------------------
// QKV projection: (B*N x 384) @ (384 x 1152) + b, scatter into q/k/vT (f16)
// Each wave: one 16-row tile x two 16-col tiles (A-fragment reused).
// grid: (1152/32, 3072/16), block: 32
// ---------------------------------------------------------------------------
__global__ void qkv_gemm_kernel(const _Float16* __restrict__ xh,   // 3072 x 384
                                const _Float16* __restrict__ whT,  // 1152 x 384
                                const float* __restrict__ qkv_b,
                                _Float16* __restrict__ q,
                                _Float16* __restrict__ k,
                                _Float16* __restrict__ vT) {
  int tn = blockIdx.x * 32;           // output column base (0..1151)
  int tm = blockIdx.y * 16;           // token-row base     (0..3071)
  v8f acc[2] = {};
#pragma unroll 3
  for (int kk = 0; kk < DIM; kk += 32) {
    v16h a  = load_a_frag_f16(xh + (size_t)tm * DIM + kk, DIM);
    v16h b0 = load_b_frag_bt(whT + (size_t)tn * DIM + kk, DIM);
    v16h b1 = load_b_frag_bt(whT + (size_t)(tn + 16) * DIM + kk, DIM);
    acc[0] = wmma_f16(a, b0, acc[0]);
    acc[1] = wmma_f16(a, b1, acc[1]);
  }
  int lane = threadIdx.x & 31;
  int hi = lane >> 4, n = lane & 15;
#pragma unroll
  for (int t = 0; t < 2; ++t) {
#pragma unroll
    for (int r = 0; r < 8; ++r) {
      int m   = r + 8 * hi;
      int row = tm + m;                // b*NTOK + token
      int col = tn + t * 16 + n;       // 0..1151
      float val = acc[t][r] + qkv_b[col];
      int which = col / DIM, rem = col % DIM;
      int h = rem >> 5, d = rem & 31;
      int b = row / NTOK, tok = row % NTOK;
      size_t hb = (size_t)(b * NHEAD + h);
      if (which == 0)       q[(hb * NTOK + tok) * HD + d] = (_Float16)(val * SCALE);
      else if (which == 1)  k[(hb * NTOK + tok) * HD + d] = (_Float16)val;
      else                  vT[(hb * HD + d) * NTOK + tok] = (_Float16)val;
    }
  }
}

// ---------------------------------------------------------------------------
// Fused flash attention + continuous position bias MLP (packed f16 MLP).
// grid: (B*H, N/64), block: 128 (4 waves, one 16-row q-tile each)
// ---------------------------------------------------------------------------
__global__ void attn_flash_kernel(const _Float16* __restrict__ q,
                                  const _Float16* __restrict__ k,
                                  const _Float16* __restrict__ vT,
                                  const float* __restrict__ coords,
                                  const float* __restrict__ cpb_w1,
                                  const float* __restrict__ cpb_b1,
                                  const float* __restrict__ cpb_w2,
                                  const float* __restrict__ cpb_b2,
                                  _Float16* __restrict__ oh) {
  __shared__ h2 w1a[CPBH], w1b[CPBH], b1v[CPBH], w2v[CPBH];  // broadcast-packed
  __shared__ __align__(32) _Float16 pstage[4][16 * 32];

  int bh = blockIdx.x;
  int b = bh / NHEAD, h = bh % NHEAD;
  int wid  = threadIdx.x >> 5;
  int lane = threadIdx.x & 31;
  int qt = (blockIdx.y * 4 + wid) * 16;
  int hiL = lane >> 4, nL = lane & 15;

  if (threadIdx.x < CPBH) {
    int c = threadIdx.x;
    w1a[c] = splat2(cpb_w1[c]);            // w1[0][c]
    w1b[c] = splat2(cpb_w1[CPBH + c]);     // w1[1][c]
    b1v[c] = splat2(cpb_b1[c]);
    w2v[c] = splat2(cpb_w2[c * NHEAD + h]);
  }
  __syncthreads();
  float b2h = cpb_b2[h];
  const h2 hzero = { (_Float16)0.0f, (_Float16)0.0f };

  // resident Q fragment and per-lane query coords
  const _Float16* qbase = q + ((size_t)bh * NTOK + qt) * HD;
  v16h qfrag = load_a_frag_f16(qbase, HD);
  float qx[8], qy[8];
#pragma unroll
  for (int r = 0; r < 8; ++r) {
    int i = qt + r + 8 * hiL;
    qx[r] = coords[((size_t)b * NTOK + i) * 2 + 0];
    qy[r] = coords[((size_t)b * NTOK + i) * 2 + 1];
  }

  float mrow[8], lrow[8];
  v8f accd0 = {}, accd1 = {};
#pragma unroll
  for (int r = 0; r < 8; ++r) { mrow[r] = -1e30f; lrow[r] = 0.0f; }

  for (int kt = 0; kt < NTOK; kt += 32) {
    const _Float16* kbase = k + ((size_t)bh * NTOK + kt) * HD;
    __builtin_prefetch(kbase + 32 * HD, 0, 1);
    v16h kb0 = load_b_frag_bt(kbase, HD);              // keys kt..kt+15
    v16h kb1 = load_b_frag_bt(kbase + 16 * HD, HD);    // keys kt+16..kt+31
    v8f zero = {};
    v8f s0 = wmma_f16(qfrag, kb0, zero);
    v8f s1 = wmma_f16(qfrag, kb1, zero);

    // key coords for this lane's columns
    int j0 = kt + nL, j1 = kt + 16 + nL;
    float kx0 = coords[((size_t)b * NTOK + j0) * 2 + 0];
    float ky0 = coords[((size_t)b * NTOK + j0) * 2 + 1];
    float kx1 = coords[((size_t)b * NTOK + j1) * 2 + 0];
    float ky1 = coords[((size_t)b * NTOK + j1) * 2 + 1];

#pragma unroll
    for (int r = 0; r < 8; ++r) {
      // --- packed cpb bias for pairs (i,j0) | (i,j1) ---
      float rx0 = qx[r] - kx0, ry0 = qy[r] - ky0;
      float rx1 = qx[r] - kx1, ry1 = qy[r] - ky1;
      float i0 = 1.0f / (sqrtf(rx0 * rx0 + ry0 * ry0) + 1e-6f);
      float i1 = 1.0f / (sqrtf(rx1 * rx1 + ry1 * ry1) + 1e-6f);
      h2 rx2 = { (_Float16)(rx0 * i0), (_Float16)(rx1 * i1) };
      h2 ry2 = { (_Float16)(ry0 * i0), (_Float16)(ry1 * i1) };
      h2 bias2 = splat2(b2h);
#pragma unroll
      for (int c = 0; c < CPBH; ++c) {
        h2 t = rx2 * w1a[c] + (ry2 * w1b[c] + b1v[c]);   // v_pk_fma_f16 x2
        t = __builtin_elementwise_max(t, hzero);          // v_pk_max_f16
        bias2 = bias2 + t * w2v[c];                       // v_pk_fma_f16
      }
      s0[r] += (float)bias2.x;
      s1[r] += (float)bias2.y;

      // --- online softmax update for row r ---
      float mx = fmaxf(s0[r], s1[r]);
      mx = fmaxf(mx, __shfl_xor(mx, 1, 32));
      mx = fmaxf(mx, __shfl_xor(mx, 2, 32));
      mx = fmaxf(mx, __shfl_xor(mx, 4, 32));
      mx = fmaxf(mx, __shfl_xor(mx, 8, 32));
      float mnew  = fmaxf(mrow[r], mx);
      float corr  = __expf(mrow[r] - mnew);
      mrow[r] = mnew;
      float p0 = __expf(s0[r] - mnew);
      float p1 = __expf(s1[r] - mnew);
      float ps = p0 + p1;
      ps += __shfl_xor(ps, 1, 32);
      ps += __shfl_xor(ps, 2, 32);
      ps += __shfl_xor(ps, 4, 32);
      ps += __shfl_xor(ps, 8, 32);
      lrow[r]  = lrow[r] * corr + ps;
      accd0[r] *= corr;
      accd1[r] *= corr;
      s0[r] = p0; s1[r] = p1;
    }

    // re-layout P: C-fragment -> LDS tile (wave-private) -> A-fragment (f16)
    _Float16* ps = pstage[wid];
#pragma unroll
    for (int r = 0; r < 8; ++r) {
      int m = r + 8 * hiL;
      ps[m * 32 + nL]      = (_Float16)s0[r];
      ps[m * 32 + 16 + nL] = (_Float16)s1[r];
    }
    asm volatile("s_wait_dscnt 0x0" ::: "memory");
    v16h pa = load_a_frag_f16(ps, 32);

    const _Float16* vbase = vT + (size_t)bh * HD * NTOK + kt;
    v16h vb0 = load_b_frag_bt(vbase, NTOK);             // d = 0..15
    v16h vb1 = load_b_frag_bt(vbase + 16 * NTOK, NTOK); // d = 16..31
    accd0 = wmma_f16(pa, vb0, accd0);
    accd1 = wmma_f16(pa, vb1, accd1);
  }

  // epilogue: normalize, write f16 rows for the projection GEMM
#pragma unroll
  for (int r = 0; r < 8; ++r) {
    int m = r + 8 * hiL;
    float inv = 1.0f / lrow[r];
    size_t orow = ((size_t)b * NTOK + qt + m) * DIM + h * HD;
    oh[orow + nL]      = (_Float16)(accd0[r] * inv);
    oh[orow + 16 + nL] = (_Float16)(accd1[r] * inv);
  }
}

// ---------------------------------------------------------------------------
// Output projection: (3072 x 384) @ (384 x 384) + b -> f32 d_out
// Each wave: one 16-row tile x two 16-col tiles.
// grid: (384/32, 3072/16), block: 32
// ---------------------------------------------------------------------------
__global__ void proj_gemm_kernel(const _Float16* __restrict__ oh,
                                 const _Float16* __restrict__ wpT,  // 384 x 384 (W^T)
                                 const float* __restrict__ proj_b,
                                 float* __restrict__ out) {
  int tn = blockIdx.x * 32;
  int tm = blockIdx.y * 16;
  v8f acc[2] = {};
#pragma unroll 3
  for (int kk = 0; kk < DIM; kk += 32) {
    v16h a  = load_a_frag_f16(oh + (size_t)tm * DIM + kk, DIM);
    v16h b0 = load_b_frag_bt(wpT + (size_t)tn * DIM + kk, DIM);
    v16h b1 = load_b_frag_bt(wpT + (size_t)(tn + 16) * DIM + kk, DIM);
    acc[0] = wmma_f16(a, b0, acc[0]);
    acc[1] = wmma_f16(a, b1, acc[1]);
  }
  int lane = threadIdx.x & 31;
  int hi = lane >> 4, n = lane & 15;
#pragma unroll
  for (int t = 0; t < 2; ++t) {
#pragma unroll
    for (int r = 0; r < 8; ++r) {
      int m = r + 8 * hi;
      int col = tn + t * 16 + n;
      out[(size_t)(tm + m) * DIM + col] = acc[t][r] + proj_b[col];
    }
  }
}

// ---------------------------------------------------------------------------
extern "C" void kernel_launch(void* const* d_in, const int* in_sizes, int n_in,
                              void* d_out, int out_size, void* d_ws, size_t ws_size,
                              hipStream_t stream) {
  const float* x      = (const float*)d_in[0];
  const float* coords = (const float*)d_in[1];
  const float* qkv_w  = (const float*)d_in[2];
  const float* qkv_b  = (const float*)d_in[3];
  const float* cpb_w1 = (const float*)d_in[4];
  const float* cpb_b1 = (const float*)d_in[5];
  const float* cpb_w2 = (const float*)d_in[6];
  const float* cpb_b2 = (const float*)d_in[7];
  const float* proj_w = (const float*)d_in[8];
  const float* proj_b = (const float*)d_in[9];
  float* out = (float*)d_out;

  // workspace carve-out (all 256B aligned)
  char* ws = (char*)d_ws;
  size_t off = 0;
  auto carve = [&](size_t bytes) {
    char* p = ws + off;
    off += (bytes + 255) & ~(size_t)255;
    return p;
  };
  _Float16* wqkvT  = (_Float16*)carve((size_t)DIM * QKVN * 2);   // 1152 x 384
  _Float16* wprojT = (_Float16*)carve((size_t)DIM * DIM * 2);    // 384 x 384
  _Float16* xh     = (_Float16*)carve((size_t)NBATCH * NTOK * DIM * 2);
  const size_t perHeadQK = (size_t)NBATCH * NHEAD * NTOK * HD;
  _Float16* qh  = (_Float16*)carve(perHeadQK * 2);
  _Float16* kh  = (_Float16*)carve(perHeadQK * 2);
  _Float16* vTh = (_Float16*)carve(perHeadQK * 2);
  _Float16* oh  = (_Float16*)carve((size_t)NBATCH * NTOK * DIM * 2);

  // 1) one-time f16 conversions: transposed weights + activations
  {
    int n1 = DIM * QKVN;
    cvt_transpose_f16_kernel<<<(n1 + 255) / 256, 256, 0, stream>>>(qkv_w, wqkvT, DIM, QKVN);
    int n2 = DIM * DIM;
    cvt_transpose_f16_kernel<<<(n2 + 255) / 256, 256, 0, stream>>>(proj_w, wprojT, DIM, DIM);
    int n3 = NBATCH * NTOK * DIM;
    cvt_f16_kernel<<<(n3 + 255) / 256, 256, 0, stream>>>(x, xh, n3);
  }
  // 2) qkv projection GEMM + scatter
  qkv_gemm_kernel<<<dim3(QKVN / 32, (NBATCH * NTOK) / 16), 32, 0, stream>>>(
      xh, wqkvT, qkv_b, qh, kh, vTh);
  // 3) fused flash attention + cpb bias (4 q-tiles per 128-thread block)
  attn_flash_kernel<<<dim3(NBATCH * NHEAD, NTOK / 64), 128, 0, stream>>>(
      qh, kh, vTh, coords, cpb_w1, cpb_b1, cpb_w2, cpb_b2, oh);
  // 4) output projection
  proj_gemm_kernel<<<dim3(DIM / 32, (NBATCH * NTOK) / 16), 32, 0, stream>>>(
      oh, wprojT, proj_b, out);
}
